// WindowAttention_16131897164213
// MI455X (gfx1250) — matrix operand order repaired and checked by reading, not compiled, not run on hardware
//
#include <hip/hip_runtime.h>

// ---- problem constants ----
#define NHEADS   16
#define CDIM     768
#define HDIM     48
#define HDPAD    64
#define NTOK     128
#define BWIN     512
#define QKVN     2304
#define MTOT     (BWIN * NTOK)            // 65536 tokens
#define SCALE_F  0.14433756729740643f     // 48^-0.5
#define L2F      1.66096404744368f        // log2(10000)/8

typedef __bf16 bf16;
typedef __attribute__((ext_vector_type(16))) __bf16 v16bf;
typedef __attribute__((ext_vector_type(8)))  __bf16 v8bf;
typedef __attribute__((ext_vector_type(8)))  float  v8f;

static __device__ __forceinline__ bf16 f2bf(float f) {
  union { float f; unsigned u; } in; in.f = f;
  unsigned r = in.u + 0x7fffu + ((in.u >> 16) & 1u);   // round-to-nearest-even
  union { unsigned short s; bf16 b; } out;
  out.s = (unsigned short)(r >> 16);
  return out.b;
}

// A fragment (16x32 bf16 MxK). row = base of this lane's M-row (row M = lane&15).
// lane<16: halves = K k0+{0..7},{16..23}; lane>=16: +8 on the base.
static __device__ __forceinline__ v16bf frag_a(const bf16* row, int k0, int lane) {
  const bf16* p = row + k0 + ((lane >> 4) << 3);
  union { v16bf v; v8bf h[2]; } u;
  u.h[0] = *(const v8bf*)(p);
  u.h[1] = *(const v8bf*)(p + 16);
  return u.v;
}
// B fragment (32x16 bf16 KxN), supplied as rows of B^T (contiguous K per column N = lane&15).
// lane<16: halves = K k0+{0..7},{16..23}; lane>=16: +8 on the base.
static __device__ __forceinline__ v16bf frag_b(const bf16* row, int k0, int lane) {
  const bf16* p = row + k0 + ((lane >> 4) << 3);
  union { v16bf v; v8bf h[2]; } u;
  u.h[0] = *(const v8bf*)(p);
  u.h[1] = *(const v8bf*)(p + 16);
  return u.v;
}

#define WMMA_BF16(A, B, C) \
  __builtin_amdgcn_wmma_f32_16x16x32_bf16(false, (A), false, (B), (short)0, (C), false, false)

// ---------------- helper kernels ----------------
__global__ void cvt_bf16_kernel(const float* __restrict__ src, bf16* __restrict__ dst, int n) {
  int i = blockIdx.x * blockDim.x + threadIdx.x;
  int stride = gridDim.x * blockDim.x;
  for (; i < n; i += stride) dst[i] = f2bf(src[i]);
}
__global__ void fill0_bf16_kernel(bf16* __restrict__ dst, int n) {
  union { unsigned short s; bf16 b; } z; z.s = 0;
  int i = blockIdx.x * blockDim.x + threadIdx.x;
  int stride = gridDim.x * blockDim.x;
  for (; i < n; i += stride) dst[i] = z.b;
}

// ---------------- stage 1: QKV GEMM + bias + RoPE + scale + relayout ----------------
// xe: [65536][768] bf16, wq: [2304][768] bf16 (qkv_w rows = output channels)
// q_s/k_s: [B*H][128][64] bf16 (padded, rope'd, q pre-scaled); vT: [B*H][48][128] bf16
__global__ __launch_bounds__(256) void qkv_rope_kernel(
    const bf16* __restrict__ xe, const bf16* __restrict__ wq,
    const float* __restrict__ qkv_b,
    bf16* __restrict__ q_s, bf16* __restrict__ k_s, bf16* __restrict__ vT)
{
  const int lane = threadIdx.x & 31;
  const int wave = threadIdx.x >> 5;
  const int N0 = blockIdx.x * 128;            // output-channel tile base (of 2304)
  const int M0 = blockIdx.y * 128 + wave * 16; // token tile base (of 65536)

  const v8f vzero = {0.f,0.f,0.f,0.f,0.f,0.f,0.f,0.f};
  v8f acc[8];
#pragma unroll
  for (int t = 0; t < 8; ++t) acc[t] = vzero;

  const bf16* arow = xe + (size_t)(M0 + (lane & 15)) * CDIM;
  for (int k0 = 0; k0 < CDIM; k0 += 32) {
    v16bf a = frag_a(arow, k0, lane);
#pragma unroll
    for (int t = 0; t < 8; ++t) {
      const bf16* brow = wq + (size_t)(N0 + t * 16 + (lane & 15)) * CDIM;
      acc[t] = WMMA_BF16(a, frag_b(brow, k0, lane), acc[t]);
    }
  }

  const int halfsel = lane >> 4;
  const float sgn  = (lane & 8) ? 1.f : -1.f;                 // rot = [-x2, x1]
  const float invf = exp2f(-(float)(lane & 7) * L2F);         // 10000^{-(d&7)/8}

#pragma unroll
  for (int t = 0; t < 8; ++t) {
    int col  = N0 + t * 16 + (lane & 15);      // global qkv channel
    int sect = col / CDIM;                     // 0=q,1=k,2=v (uniform per tile)
    int c    = col - sect * CDIM;
    int head = c / HDIM;                       // uniform per tile
    int f    = c - head * HDIM;                // 0..47
    int chunk = f >> 4;                        // 0:z 1:h 2:w (uniform per tile)
    float biasv = qkv_b[col];

#pragma unroll
    for (int j = 0; j < 8; ++j) {
      int token = M0 + j + 8 * halfsel;
      int b_ = token >> 7;
      int nt = token & 127;                    // token in window: z*64 + h*8 + w
      float val = acc[t][j] + biasv;
      if (sect == 2) {
        vT[((size_t)(b_ * NHEADS + head) * HDIM + f) * NTOK + nt] = f2bf(val);
      } else {
        int z = nt >> 6, hh = (nt >> 3) & 7, wc = nt & 7;
        float pos = (chunk == 0) ? (float)z : (chunk == 1) ? (float)hh : (float)wc;
        float ang = pos * invf;
        float cv = __cosf(ang), sv = __sinf(ang);
        float partner = __shfl_xor(val, 8, 32); // pairs d <-> d^8 within chunk
        float r = val * cv + sgn * partner * sv;
        if (sect == 0) {
          r *= SCALE_F;
          q_s[((size_t)(b_ * NHEADS + head) * NTOK + nt) * HDPAD + f] = f2bf(r);
        } else {
          k_s[((size_t)(b_ * NHEADS + head) * NTOK + nt) * HDPAD + f] = f2bf(r);
        }
      }
    }
  }
}

// ---------------- stage 2: attention per (batch, head) ----------------
// S = Q@K^T (+mask), softmax, O = P@V ; one block per (b,h), wave w owns query rows 16w..16w+15
__global__ __launch_bounds__(256) void attn_kernel(
    const bf16* __restrict__ q_s, const bf16* __restrict__ k_s,
    const bf16* __restrict__ vT, const float* __restrict__ mask,
    bf16* __restrict__ o_s)
{
  __shared__ __align__(16) bf16 lp[8][16][128];   // P tiles, re-staged as A fragments

  const int lane = threadIdx.x & 31;
  const int wave = threadIdx.x >> 5;
  const int bh   = blockIdx.x;
  const int b_   = bh >> 4;
  const int head = bh & 15;
  const int widx = b_ & 63;
  const int halfsel = lane >> 4;

  const bf16* qbase = q_s + (size_t)bh * NTOK * HDPAD;
  const bf16* kbase = k_s + (size_t)bh * NTOK * HDPAD;
  const bf16* vbase = vT  + (size_t)bh * HDIM * NTOK;

  const bf16* qrow = qbase + (size_t)(wave * 16 + (lane & 15)) * HDPAD;
  v16bf a0 = frag_a(qrow, 0, lane);
  v16bf a1 = frag_a(qrow, 32, lane);

  const v8f vzero = {0.f,0.f,0.f,0.f,0.f,0.f,0.f,0.f};
  v8f s[8];
#pragma unroll
  for (int t = 0; t < 8; ++t) {
    const bf16* krow = kbase + (size_t)(t * 16 + (lane & 15)) * HDPAD;
    v8f z = WMMA_BF16(a0, frag_b(krow, 0, lane), vzero);
    s[t]  = WMMA_BF16(a1, frag_b(krow, 32, lane), z);
  }

  // + mask
#pragma unroll
  for (int t = 0; t < 8; ++t)
#pragma unroll
    for (int j = 0; j < 8; ++j) {
      int m = wave * 16 + j + 8 * halfsel;
      int n = t * 16 + (lane & 15);
      s[t][j] += mask[((size_t)widx * NTOK + m) * NTOK + n];
    }

  // softmax per row (row r=j+8*halfsel lives in vgpr j across 16 lanes of this half)
#pragma unroll
  for (int j = 0; j < 8; ++j) {
    float mx = s[0][j];
#pragma unroll
    for (int t = 1; t < 8; ++t) mx = fmaxf(mx, s[t][j]);
    mx = fmaxf(mx, __shfl_xor(mx, 1, 32));
    mx = fmaxf(mx, __shfl_xor(mx, 2, 32));
    mx = fmaxf(mx, __shfl_xor(mx, 4, 32));
    mx = fmaxf(mx, __shfl_xor(mx, 8, 32));
    float sum = 0.f;
#pragma unroll
    for (int t = 0; t < 8; ++t) { float e = __expf(s[t][j] - mx); s[t][j] = e; sum += e; }
    sum += __shfl_xor(sum, 1, 32);
    sum += __shfl_xor(sum, 2, 32);
    sum += __shfl_xor(sum, 4, 32);
    sum += __shfl_xor(sum, 8, 32);
    float inv = 1.f / sum;
#pragma unroll
    for (int t = 0; t < 8; ++t) s[t][j] *= inv;
  }

  // re-stage P into LDS (row-major 16x128 per wave) for A-fragment reads
#pragma unroll
  for (int t = 0; t < 8; ++t)
#pragma unroll
    for (int j = 0; j < 8; ++j)
      lp[wave][j + 8 * halfsel][t * 16 + (lane & 15)] = f2bf(s[t][j]);
  __syncthreads();

  // O = P @ V  (V stored transposed: vT[f][tok] -> B-fragment rows are contiguous)
  v8f o0 = vzero, o1 = vzero, o2 = vzero;
  const bf16* prow = &lp[wave][lane & 15][0];
#pragma unroll
  for (int kk = 0; kk < NTOK; kk += 32) {
    v16bf pa = frag_a(prow, kk, lane);
    o0 = WMMA_BF16(pa, frag_b(vbase + (size_t)(0  + (lane & 15)) * NTOK, kk, lane), o0);
    o1 = WMMA_BF16(pa, frag_b(vbase + (size_t)(16 + (lane & 15)) * NTOK, kk, lane), o1);
    o2 = WMMA_BF16(pa, frag_b(vbase + (size_t)(32 + (lane & 15)) * NTOK, kk, lane), o2);
  }

  // scatter to o_s[token][768] with channel = head*48 + f (matches transpose(0,2,1,3))
#pragma unroll
  for (int ft = 0; ft < 3; ++ft) {
    const v8f& ov = (ft == 0) ? o0 : (ft == 1) ? o1 : o2;
    int ch = head * HDIM + ft * 16 + (lane & 15);
#pragma unroll
    for (int j = 0; j < 8; ++j) {
      int tok = b_ * NTOK + wave * 16 + j + 8 * halfsel;
      o_s[(size_t)tok * CDIM + ch] = f2bf(ov[j]);
    }
  }
}

// ---------------- stage 3: output projection ----------------
__global__ __launch_bounds__(256) void proj_kernel(
    const bf16* __restrict__ o_s, const bf16* __restrict__ wp,
    const float* __restrict__ proj_b, float* __restrict__ out)
{
  const int lane = threadIdx.x & 31;
  const int wave = threadIdx.x >> 5;
  const int N0 = blockIdx.x * 128;
  const int M0 = blockIdx.y * 128 + wave * 16;

  const v8f vzero = {0.f,0.f,0.f,0.f,0.f,0.f,0.f,0.f};
  v8f acc[8];
#pragma unroll
  for (int t = 0; t < 8; ++t) acc[t] = vzero;

  const bf16* arow = o_s + (size_t)(M0 + (lane & 15)) * CDIM;
  for (int k0 = 0; k0 < CDIM; k0 += 32) {
    v16bf a = frag_a(arow, k0, lane);
#pragma unroll
    for (int t = 0; t < 8; ++t) {
      const bf16* brow = wp + (size_t)(N0 + t * 16 + (lane & 15)) * CDIM;
      acc[t] = WMMA_BF16(a, frag_b(brow, k0, lane), acc[t]);
    }
  }

  const int halfsel = lane >> 4;
#pragma unroll
  for (int t = 0; t < 8; ++t) {
    int col = N0 + t * 16 + (lane & 15);
    float biasv = proj_b[col];
#pragma unroll
    for (int j = 0; j < 8; ++j) {
      int token = M0 + j + 8 * halfsel;
      out[(size_t)token * CDIM + col] = acc[t][j] + biasv;
    }
  }
}

// ---------------- launch ----------------
extern "C" void kernel_launch(void* const* d_in, const int* in_sizes, int n_in,
                              void* d_out, int out_size, void* d_ws, size_t ws_size,
                              hipStream_t stream) {
  const float* x      = (const float*)d_in[0];
  const float* mask   = (const float*)d_in[1];
  const float* qkv_w  = (const float*)d_in[2];
  const float* qkv_b  = (const float*)d_in[3];
  const float* proj_w = (const float*)d_in[4];
  const float* proj_b = (const float*)d_in[5];
  float* out = (float*)d_out;

  char* ws = (char*)d_ws;
  bf16* xe  = (bf16*)ws; ws += (size_t)MTOT * CDIM * sizeof(bf16);
  bf16* wq  = (bf16*)ws; ws += (size_t)QKVN * CDIM * sizeof(bf16);
  bf16* wp  = (bf16*)ws; ws += (size_t)CDIM * CDIM * sizeof(bf16);
  bf16* q_s = (bf16*)ws; ws += (size_t)BWIN * NHEADS * NTOK * HDPAD * sizeof(bf16);
  bf16* k_s = (bf16*)ws; ws += (size_t)BWIN * NHEADS * NTOK * HDPAD * sizeof(bf16);
  bf16* vT  = (bf16*)ws; ws += (size_t)BWIN * NHEADS * HDIM * NTOK * sizeof(bf16);
  bf16* o_s = (bf16*)ws; ws += (size_t)MTOT * CDIM * sizeof(bf16);

  cvt_bf16_kernel<<<4096, 256, 0, stream>>>(x, xe, MTOT * CDIM);
  cvt_bf16_kernel<<<1024, 256, 0, stream>>>(qkv_w, wq, QKVN * CDIM);
  cvt_bf16_kernel<<<512, 256, 0, stream>>>(proj_w, wp, CDIM * CDIM);
  fill0_bf16_kernel<<<2048, 256, 0, stream>>>(q_s, BWIN * NHEADS * NTOK * HDPAD);
  fill0_bf16_kernel<<<2048, 256, 0, stream>>>(k_s, BWIN * NHEADS * NTOK * HDPAD);

  qkv_rope_kernel<<<dim3(QKVN / 128, MTOT / 128), 256, 0, stream>>>(
      xe, wq, qkv_b, q_s, k_s, vT);
  attn_kernel<<<BWIN * NHEADS, 256, 0, stream>>>(q_s, k_s, vT, mask, o_s);
  proj_kernel<<<dim3(CDIM / 128, MTOT / 128), 256, 0, stream>>>(o_s, wp, proj_b, out);
}
